// BiLSTM_CRF_21517786153229
// MI455X (gfx1250) — matrix operand order repaired and checked by reading, not compiled, not run on hardware
//
#include <hip/hip_runtime.h>
#include <hip/hip_bf16.h>
#include <math.h>

#define S_LEN   2048
#define E_DIM   1024
#define H_DIM   512
#define G_DIM   2048        // 4*H (gate dim)
#define T_TAGS  16
#define START_TAG 0
#define STOP_TAG  1
#define NEG_INF  (-10000.0f)

typedef __bf16 bf16;
typedef __attribute__((ext_vector_type(16))) __bf16 v16bf;
typedef __attribute__((ext_vector_type(8)))  float  v8f;

// ---------- bf16 helpers ----------
__device__ __forceinline__ bf16 f2bf(float f) {
    unsigned u = __builtin_bit_cast(unsigned, f);
    unsigned r = (u + 0x7FFFu + ((u >> 16) & 1u)) >> 16;   // round-to-nearest-even
    unsigned short h = (unsigned short)r;
    return __builtin_bit_cast(bf16, h);
}
__device__ __forceinline__ float bf_lo(unsigned u) { return __builtin_bit_cast(float, u << 16); }
__device__ __forceinline__ float bf_hi(unsigned u) { return __builtin_bit_cast(float, u & 0xFFFF0000u); }

// ---------- elementwise f32 -> bf16 ----------
__global__ void cast_f32_bf16(const float* __restrict__ in, bf16* __restrict__ out, int n) {
    int i = blockIdx.x * blockDim.x + threadIdx.x;
    if (i < n) out[i] = f2bf(in[i]);
}

// ---------- embedding gather + cast:  x[s,e] = bf16(emb[sentence[s], e]) ----------
__global__ void gather_cast(const int* __restrict__ sent, const float* __restrict__ emb,
                            bf16* __restrict__ x) {
    int i = blockIdx.x * blockDim.x + threadIdx.x;      // over S*E
    int s = i >> 10;                                    // / E_DIM
    int e = i & (E_DIM - 1);
    x[i] = f2bf(emb[(size_t)sent[s] * E_DIM + e]);
}

// ---------- WMMA GEMM: C[S,N] = A[S,E](bf16) * B[N,E]^T(bf16) + bias1[N] + bias2[N] ----------
// One wave per 16x64 output tile: 4 accumulators share one A fragment per K-step.
// All 10 loads of a K-step are issued before its 4 WMMAs (partial s_wait_loadcnt),
// and the K loop is fully unrolled so the scheduler pipelines loads across steps.
// A layout (16-bit A 16x32): lane L / L+16 both hold row M=L.  B col n == row n of
// Wih (contiguous K): lane n holds K{0..15}, lane n+16 holds K{16..31}.
__global__ void __launch_bounds__(32)
gemm_xg(const bf16* __restrict__ A, const bf16* __restrict__ B,
        const float* __restrict__ bias1, const float* __restrict__ bias2,
        float* __restrict__ C, int N) {
    int lane = threadIdx.x;
    int half = lane >> 4;          // 0: lanes 0-15, 1: lanes 16-31
    int l    = lane & 15;
    int tm = blockIdx.y << 4;
    int tn = blockIdx.x << 6;      // 64-wide N tile
    const bf16* arow = A + (size_t)(tm + l) * E_DIM;
    const bf16* br0  = B + (size_t)(tn +  0 + l) * E_DIM;
    const bf16* br1  = B + (size_t)(tn + 16 + l) * E_DIM;
    const bf16* br2  = B + (size_t)(tn + 32 + l) * E_DIM;
    const bf16* br3  = B + (size_t)(tn + 48 + l) * E_DIM;

    v8f acc0 = (v8f){0.f,0.f,0.f,0.f,0.f,0.f,0.f,0.f};
    v8f acc1 = acc0, acc2 = acc0, acc3 = acc0;

    union AB { v16bf v; uint4 u[2]; };

    #pragma unroll
    for (int k0 = 0; k0 < E_DIM; k0 += 32) {
        AB a, b0, b1, b2, b3;
        // A fragment: two contiguous 16B chunks per lane
        a.u[0]  = *(const uint4*)(arow + k0 + half * 16);
        a.u[1]  = *(const uint4*)(arow + k0 + half * 16 + 8);
        // Four B fragments: 32 contiguous bytes per lane each
        b0.u[0] = *(const uint4*)(br0 + k0 + half * 16);
        b0.u[1] = *(const uint4*)(br0 + k0 + half * 16 + 8);
        b1.u[0] = *(const uint4*)(br1 + k0 + half * 16);
        b1.u[1] = *(const uint4*)(br1 + k0 + half * 16 + 8);
        b2.u[0] = *(const uint4*)(br2 + k0 + half * 16);
        b2.u[1] = *(const uint4*)(br2 + k0 + half * 16 + 8);
        b3.u[0] = *(const uint4*)(br3 + k0 + half * 16);
        b3.u[1] = *(const uint4*)(br3 + k0 + half * 16 + 8);
        // One 256B prefetch covers 4 K-steps; compile-time predicated under full unroll
        if ((k0 & 127) == 0) {
            __builtin_prefetch(arow + k0 + 256, 0, 1);   // global_prefetch_b8
            __builtin_prefetch(br0  + k0 + 256, 0, 1);
            __builtin_prefetch(br1  + k0 + 256, 0, 1);
            __builtin_prefetch(br2  + k0 + 256, 0, 1);
            __builtin_prefetch(br3  + k0 + 256, 0, 1);
        }
        acc0 = __builtin_amdgcn_wmma_f32_16x16x32_bf16(false, a.v, false, b0.v, (short)0, acc0, false, false);
        acc1 = __builtin_amdgcn_wmma_f32_16x16x32_bf16(false, a.v, false, b1.v, (short)0, acc1, false, false);
        acc2 = __builtin_amdgcn_wmma_f32_16x16x32_bf16(false, a.v, false, b2.v, (short)0, acc2, false, false);
        acc3 = __builtin_amdgcn_wmma_f32_16x16x32_bf16(false, a.v, false, b3.v, (short)0, acc3, false, false);
    }

    // C/D layout: VGPR r -> (M = r + half*8, N = l)
    v8f accs[4] = {acc0, acc1, acc2, acc3};
    #pragma unroll
    for (int nb = 0; nb < 4; ++nb) {
        int col = tn + nb * 16 + l;
        float bb = bias1[col] + bias2[col];
        #pragma unroll
        for (int r = 0; r < 8; ++r) {
            int row = tm + r + half * 8;
            C[(size_t)row * N + col] = accs[nb][r] + bb;
        }
    }
}

// ---------- sequential LSTM, one block per direction ----------
// blockDim = 1024: each thread produces 2 of the 2048 gate pre-activations per step
// (512-MAC bf16 dot each); h (512 f32) and gates (2048 f32) staged in LDS.
__global__ void __launch_bounds__(1024)
lstm_seq(const float* __restrict__ xg_f, const float* __restrict__ xg_b,
         const bf16* __restrict__ whhf, const bf16* __restrict__ whhb,
         const float* __restrict__ h0,  const float* __restrict__ c0,
         float* __restrict__ hs) {
    __shared__ float sh[H_DIM];
    __shared__ float sg[G_DIM];
    int dir = blockIdx.x;
    int tid = threadIdx.x;
    const float* xg  = dir ? xg_b : xg_f;
    const bf16*  whh = dir ? whhb : whhf;

    if (tid < H_DIM) sh[tid] = h0[dir * H_DIM + tid];
    float c = (tid < H_DIM) ? c0[dir * H_DIM + tid] : 0.f;
    __syncthreads();

    const bf16* w0 = whh + (size_t)(2 * tid) * H_DIM;
    const bf16* w1 = w0 + H_DIM;

    for (int step = 0; step < S_LEN; ++step) {
        int t = dir ? (S_LEN - 1 - step) : step;   // backward dir walks reversed time
        const float* xr = xg + (size_t)t * G_DIM;

        float a0 = 0.f, a1 = 0.f;
        #pragma unroll 4
        for (int k = 0; k < H_DIM; k += 8) {
            uint4 u = *(const uint4*)(w0 + k);
            uint4 v = *(const uint4*)(w1 + k);
            float s0 = sh[k+0], s1 = sh[k+1], s2 = sh[k+2], s3 = sh[k+3];
            float s4 = sh[k+4], s5 = sh[k+5], s6 = sh[k+6], s7 = sh[k+7];
            a0 += bf_lo(u.x)*s0 + bf_hi(u.x)*s1 + bf_lo(u.y)*s2 + bf_hi(u.y)*s3
                + bf_lo(u.z)*s4 + bf_hi(u.z)*s5 + bf_lo(u.w)*s6 + bf_hi(u.w)*s7;
            a1 += bf_lo(v.x)*s0 + bf_hi(v.x)*s1 + bf_lo(v.y)*s2 + bf_hi(v.y)*s3
                + bf_lo(v.z)*s4 + bf_hi(v.z)*s5 + bf_lo(v.w)*s6 + bf_hi(v.w)*s7;
        }
        sg[2*tid]     = xr[2*tid]     + a0;
        sg[2*tid + 1] = xr[2*tid + 1] + a1;
        __syncthreads();

        if (tid < H_DIM) {
            float ig = 1.f / (1.f + __expf(-sg[tid]));
            float fg = 1.f / (1.f + __expf(-sg[H_DIM   + tid]));
            float gg = tanhf(sg[2*H_DIM + tid]);
            float og = 1.f / (1.f + __expf(-sg[3*H_DIM + tid]));
            c = fg * c + ig * gg;
            float h = og * tanhf(c);
            sh[tid] = h;
            hs[(size_t)t * (2 * H_DIM) + dir * H_DIM + tid] = h;   // hf | hb concat
        }
        __syncthreads();
    }
}

// ---------- feats = hs[S,2H] @ Wout[T,2H]^T + bout ----------
__global__ void feats_proj(const float* __restrict__ hs, const float* __restrict__ wout,
                           const float* __restrict__ bout, float* __restrict__ feats) {
    int i = blockIdx.x * blockDim.x + threadIdx.x;     // over S*T
    int s = i >> 4, tag = i & 15;
    const float4* hr = (const float4*)(hs   + (size_t)s   * (2 * H_DIM));
    const float4* wr = (const float4*)(wout + (size_t)tag * (2 * H_DIM));
    float acc = 0.f;
    for (int k = 0; k < (2 * H_DIM) / 4; ++k) {
        float4 a = hr[k], b = wr[k];
        acc += a.x*b.x + a.y*b.y + a.z*b.z + a.w*b.w;
    }
    feats[i] = acc + bout[tag];
}

// ---------- Viterbi scan + backtrace (single wave) ----------
__global__ void __launch_bounds__(32)
viterbi(const float* __restrict__ feats, const float* __restrict__ trans,
        int* __restrict__ bps, float* __restrict__ out) {
    __shared__ float v[T_TAGS], vn[T_TAGS], tr[T_TAGS * T_TAGS];
    int tid = threadIdx.x;
    for (int i = tid; i < T_TAGS * T_TAGS; i += 32) tr[i] = trans[i];
    if (tid < T_TAGS) v[tid] = (tid == START_TAG) ? 0.f : NEG_INF;
    __syncthreads();

    for (int s = 0; s < S_LEN; ++s) {
        if (tid < T_TAGS) {
            float best = v[0] + tr[tid * T_TAGS + 0];
            int bi = 0;
            #pragma unroll
            for (int p = 1; p < T_TAGS; ++p) {
                float sc = v[p] + tr[tid * T_TAGS + p];
                if (sc > best) { best = sc; bi = p; }   // first-max like jnp.argmax
            }
            bps[s * T_TAGS + tid] = bi;
            vn[tid] = best + feats[s * T_TAGS + tid];
        }
        __syncthreads();
        if (tid < T_TAGS) v[tid] = vn[tid];
        __syncthreads();
    }

    if (tid == 0) {
        float best = v[0] + tr[STOP_TAG * T_TAGS + 0];
        int bt = 0;
        for (int p = 1; p < T_TAGS; ++p) {
            float sc = v[p] + tr[STOP_TAG * T_TAGS + p];
            if (sc > best) { best = sc; bt = p; }
        }
        out[0] = best;                               // path_score
        int tag = bt;
        for (int s = S_LEN - 1; s >= 0; --s) {       // reverse scan: path[s]=carry; carry=bp[s][carry]
            out[1 + s] = (float)tag;
            tag = bps[s * T_TAGS + tag];
        }
    }
}

extern "C" void kernel_launch(void* const* d_in, const int* in_sizes, int n_in,
                              void* d_out, int out_size, void* d_ws, size_t ws_size,
                              hipStream_t stream) {
    (void)in_sizes; (void)n_in; (void)out_size; (void)ws_size;
    const int*   sent  = (const int*)d_in[0];
    const float* emb   = (const float*)d_in[1];
    const float* wih_f = (const float*)d_in[2];
    const float* whh_f = (const float*)d_in[3];
    const float* bih_f = (const float*)d_in[4];
    const float* bhh_f = (const float*)d_in[5];
    const float* wih_b = (const float*)d_in[6];
    const float* whh_b = (const float*)d_in[7];
    const float* bih_b = (const float*)d_in[8];
    const float* bhh_b = (const float*)d_in[9];
    const float* h0    = (const float*)d_in[10];
    const float* c0    = (const float*)d_in[11];
    const float* wout  = (const float*)d_in[12];
    const float* bout  = (const float*)d_in[13];
    const float* trans = (const float*)d_in[14];
    float* out = (float*)d_out;

    // Workspace carve-out (~58 MB), 256B aligned
    char* w = (char*)d_ws;
    auto alloc = [&](size_t bytes) { char* p = w; w += (bytes + 255) & ~(size_t)255; return p; };
    bf16*  x_bf    = (bf16*) alloc((size_t)S_LEN * E_DIM * 2);
    bf16*  wihf_bf = (bf16*) alloc((size_t)G_DIM * E_DIM * 2);
    bf16*  wihb_bf = (bf16*) alloc((size_t)G_DIM * E_DIM * 2);
    bf16*  whhf_bf = (bf16*) alloc((size_t)G_DIM * H_DIM * 2);
    bf16*  whhb_bf = (bf16*) alloc((size_t)G_DIM * H_DIM * 2);
    float* xg_f    = (float*)alloc((size_t)S_LEN * G_DIM * 4);
    float* xg_b    = (float*)alloc((size_t)S_LEN * G_DIM * 4);
    float* hs      = (float*)alloc((size_t)S_LEN * 2 * H_DIM * 4);
    float* feats   = (float*)alloc((size_t)S_LEN * T_TAGS * 4);
    int*   bps     = (int*)  alloc((size_t)S_LEN * T_TAGS * 4);

    gather_cast<<<(S_LEN * E_DIM) / 256, 256, 0, stream>>>(sent, emb, x_bf);
    cast_f32_bf16<<<(G_DIM * E_DIM) / 256, 256, 0, stream>>>(wih_f, wihf_bf, G_DIM * E_DIM);
    cast_f32_bf16<<<(G_DIM * E_DIM) / 256, 256, 0, stream>>>(wih_b, wihb_bf, G_DIM * E_DIM);
    cast_f32_bf16<<<(G_DIM * H_DIM) / 256, 256, 0, stream>>>(whh_f, whhf_bf, G_DIM * H_DIM);
    cast_f32_bf16<<<(G_DIM * H_DIM) / 256, 256, 0, stream>>>(whh_b, whhb_bf, G_DIM * H_DIM);

    dim3 gg(G_DIM / 64, S_LEN / 16);     // 32 x 128 tiles of 16x64, one wave each
    gemm_xg<<<gg, 32, 0, stream>>>(x_bf, wihf_bf, bih_f, bhh_f, xg_f, G_DIM);
    gemm_xg<<<gg, 32, 0, stream>>>(x_bf, wihb_bf, bih_b, bhh_b, xg_b, G_DIM);

    lstm_seq<<<2, 1024, 0, stream>>>(xg_f, xg_b, whhf_bf, whhb_bf, h0, c0, hs);
    feats_proj<<<(S_LEN * T_TAGS) / 256, 256, 0, stream>>>(hs, wout, bout, feats);
    viterbi<<<1, 32, 0, stream>>>(feats, trans, bps, out);
}
